// WESREnhancedAttention_6055903887982
// MI455X (gfx1250) — matrix operand order repaired and checked by reading, not compiled, not run on hardware
//
#include <hip/hip_runtime.h>
#include <hip/hip_bf16.h>
#include <math.h>

// ---------------- problem constants ----------------
#define HIDDEN 1024
#define HEADS  16
#define HDIM   64
#define BATCH  2
#define SEQ    2048
#define ROWSZ  (BATCH * SEQ)     // 4096
#define WESR   7

// ---------------- WMMA tile config ----------------
#define BM 128                   // block tile M
#define BN 64                    // block tile N
#define BK 32                    // k-step
#define TPB 128                  // 4 waves; each wave owns 32(M) x 64(N)
#define ASTR (BK + 8)            // padded LDS row stride in halves (80B, 16B-aligned)
#define BSTR (BK + 8)

typedef __attribute__((ext_vector_type(16))) _Float16 v16h;
typedef __attribute__((ext_vector_type(8)))  float    v8f;

// ============ async tile staging: f16 global -> f16 LDS (byte copy) ============
// Copies ROWS_T rows x 32 halves from row-major f16 source into LDS [row][ASTR].
// CDNA5 GLOBAL_LOAD_ASYNC_TO_LDS_B128 (ASYNCcnt); 16B chunks, all 16B aligned.
template<int ROWS_T>
__device__ __forceinline__ void stage_async_h(const _Float16* __restrict__ g, long ld,
                                              int r0, int c0, _Float16* lds, int tid) {
#pragma unroll
  for (int i = 0; i < ROWS_T / 32; ++i) {          // ROWS_T*4 chunks / 128 threads
    int idx = tid + i * TPB;
    int r   = idx >> 2;                            // row
    int c8  = (idx & 3) << 3;                      // half-offset 0,8,16,24
    unsigned dst = (unsigned)(size_t)(lds + r * ASTR + c8);       // LDS byte offset
    const _Float16* src = g + (long)(r0 + r) * ld + c0 + c8;
    asm volatile("global_load_async_to_lds_b128 %0, %1, off"
                 :: "v"(dst), "v"((unsigned long long)(size_t)src)
                 : "memory");
  }
}

// wait until this wave's ASYNCcnt <= N (async completions return in order)
template<int N>
__device__ __forceinline__ void wait_async_le() {
  asm volatile("s_wait_asynccnt %0" :: "n"(N) : "memory");
}

// ============ f32 -> f16 converting A stager (for attn, which must stay f32) ====
__device__ __forceinline__ void stage_cvt_a128(const float* __restrict__ A, long lda,
                                               int m0, int k0, _Float16* Ah, int tid) {
#pragma unroll
  for (int i = 0; i < 8; ++i) {                    // 128x32 f32, float4 chunks
    int idx = tid + i * TPB;
    int r   = idx >> 3;
    int c4  = (idx & 7) << 2;
    const float4 v = *reinterpret_cast<const float4*>(A + (long)(m0 + r) * lda + k0 + c4);
    _Float16* d = Ah + r * ASTR + c4;
    d[0] = (_Float16)v.x; d[1] = (_Float16)v.y;
    d[2] = (_Float16)v.z; d[3] = (_Float16)v.w;
  }
}

// ================= fragment loads (ISA 7.12.2 layouts) =================
__device__ __forceinline__ v16h frag_a(const _Float16* Ah, int m_off, int lane) {
  // lanes 0-15: K 0-7 & 16-23 ; lanes 16-31: K 8-15 & 24-31
  const _Float16* p = Ah + (m_off + (lane & 15)) * ASTR + ((lane & 16) ? 8 : 0);
  v16h a;
#pragma unroll
  for (int i = 0; i < 8; ++i) { a[i] = p[i]; a[8 + i] = p[16 + i]; }
  return a;
}

__device__ __forceinline__ v16h frag_b(const _Float16* Bh, int n_off, int lane) {
  // Bh stored [n][k]; lanes 0-15: K 0-7 & 16-23 ; lanes 16-31: K 8-15 & 24-31
  const _Float16* p = Bh + (n_off + (lane & 15)) * BSTR + ((lane & 16) ? 8 : 0);
  v16h b;
#pragma unroll
  for (int i = 0; i < 8; ++i) { b[i] = p[i]; b[8 + i] = p[16 + i]; }
  return b;
}

// 2(M) x 4(N) WMMA subtiles per wave per k-step
__device__ __forceinline__ void mma_2x4(const _Float16* Ah, const _Float16* Bh,
                                        int wm, int lane, v8f acc[2][4]) {
  v16h a0 = frag_a(Ah, wm,      lane);
  v16h a1 = frag_a(Ah, wm + 16, lane);
#pragma unroll
  for (int j = 0; j < 4; ++j) {
    v16h b = frag_b(Bh, j * 16, lane);
    acc[0][j] = __builtin_amdgcn_wmma_f32_16x16x32_f16(false, a0, false, b, (short)0, acc[0][j], false, false);
    acc[1][j] = __builtin_amdgcn_wmma_f32_16x16x32_f16(false, a1, false, b, (short)0, acc[1][j], false, false);
  }
}

// ===== generic GEMM: C[M,N] = A[M,K](f16) @ Bt[N,K](f16)^T + bias, C f32 or f16 =====
// Double-buffered: async copies for tile k+1 overlap WMMA on tile k.
template<bool HALF_OUT>
__global__ __launch_bounds__(TPB) void gemm_nt_kernel(
    const _Float16* __restrict__ A, long lda,
    const _Float16* __restrict__ Bt, long ldb,
    const float* __restrict__ bias,
    void* __restrict__ C, long ldc, int Kdim) {
  __shared__ __align__(16) _Float16 Ah[2][BM * ASTR];
  __shared__ __align__(16) _Float16 Bh[2][BN * BSTR];
  const int tid = threadIdx.x, lane = tid & 31, wid = tid >> 5;
  const int wm = wid * 32;
  const int m0 = blockIdx.y * BM, n0 = blockIdx.x * BN;

  v8f acc[2][4] = {};
  stage_async_h<BM>(A,  lda, m0, 0, Ah[0], tid);
  stage_async_h<BN>(Bt, ldb, n0, 0, Bh[0], tid);
  int cur = 0;
  for (int k0 = 0; k0 < Kdim; k0 += BK) {
    __syncthreads();                               // all waves done reading buf[cur^1]
    if (k0 + BK < Kdim) {
      stage_async_h<BM>(A,  lda, m0, k0 + BK, Ah[cur ^ 1], tid);
      stage_async_h<BN>(Bt, ldb, n0, k0 + BK, Bh[cur ^ 1], tid);
      wait_async_le<6>();                          // tile k landed; k+1 in flight
    } else {
      wait_async_le<0>();
    }
    __syncthreads();                               // buf[cur] visible to all waves
    mma_2x4(Ah[cur], Bh[cur], wm, lane, acc);
    cur ^= 1;
  }

  const int lane15 = lane & 15;
  const int rbase  = m0 + wm + ((lane & 16) ? 8 : 0);
  float bv[4];
#pragma unroll
  for (int j = 0; j < 4; ++j) bv[j] = bias[n0 + j * 16 + lane15];
#pragma unroll
  for (int i = 0; i < 2; ++i)
#pragma unroll
    for (int r = 0; r < 8; ++r) {
      long row = rbase + i * 16 + r;
#pragma unroll
      for (int j = 0; j < 4; ++j) {
        float val = acc[i][j][r] + bv[j];
        if (HALF_OUT) ((_Float16*)C)[row * ldc + n0 + j * 16 + lane15] = (_Float16)val;
        else          ((float*)C)   [row * ldc + n0 + j * 16 + lane15] = val;
      }
    }
}

// ===== scores: S[b,h,q,k] = mask ? gate*(Q.K^T)/8 : -inf (pre-softmax, f32) =====
__global__ __launch_bounds__(TPB) void scores_kernel(
    const _Float16* __restrict__ qh, const _Float16* __restrict__ kh,
    const float* __restrict__ gate, const int* __restrict__ amask,
    float* __restrict__ attn) {
  __shared__ __align__(16) _Float16 Ah[2][BM * ASTR];
  __shared__ __align__(16) _Float16 Bh[2][BN * BSTR];
  const int bh = blockIdx.z, b = bh >> 4, h = bh & 15;
  const _Float16* A  = qh + (long)b * SEQ * HIDDEN + h * HDIM;   // [SEQ,64] ld=HIDDEN
  const _Float16* Bt = kh + (long)b * SEQ * HIDDEN + h * HDIM;   // key rows -> B^T rows
  float* S = attn + (long)bh * SEQ * SEQ;

  const int tid = threadIdx.x, lane = tid & 31, wid = tid >> 5;
  const int wm = wid * 32;
  const int m0 = blockIdx.y * BM, n0 = blockIdx.x * BN;

  v8f acc[2][4] = {};
  stage_async_h<BM>(A,  HIDDEN, m0, 0, Ah[0], tid);
  stage_async_h<BN>(Bt, HIDDEN, n0, 0, Bh[0], tid);
  int cur = 0;
#pragma unroll
  for (int k0 = 0; k0 < HDIM; k0 += BK) {
    __syncthreads();
    if (k0 + BK < HDIM) {
      stage_async_h<BM>(A,  HIDDEN, m0, k0 + BK, Ah[cur ^ 1], tid);
      stage_async_h<BN>(Bt, HIDDEN, n0, k0 + BK, Bh[cur ^ 1], tid);
      wait_async_le<6>();
    } else {
      wait_async_le<0>();
    }
    __syncthreads();
    mma_2x4(Ah[cur], Bh[cur], wm, lane, acc);
    cur ^= 1;
  }

  const int lane15 = lane & 15;
  int mk[4];
#pragma unroll
  for (int j = 0; j < 4; ++j) mk[j] = amask[b * SEQ + n0 + j * 16 + lane15];
  const int rbase = m0 + wm + ((lane & 16) ? 8 : 0);
#pragma unroll
  for (int i = 0; i < 2; ++i)
#pragma unroll
    for (int r = 0; r < 8; ++r) {
      int row = rbase + i * 16 + r;
      float g = 0.125f * gate[((long)b * SEQ + row) * HEADS + h];
      float* sp = S + (long)row * SEQ + n0;
#pragma unroll
      for (int j = 0; j < 4; ++j)
        sp[j * 16 + lane15] = mk[j] ? acc[i][j][r] * g : -INFINITY;
    }
}

// ===== attn(f32) @ V^T(f16 [B,H,D,L]) -> ctxh f16 [B,L,HIDDEN] =====
__global__ __launch_bounds__(TPB) void av_kernel(
    const float* __restrict__ attn, const _Float16* __restrict__ vT,
    _Float16* __restrict__ ctxh) {
  __shared__ __align__(16) _Float16 Ah[2][BM * ASTR];
  __shared__ __align__(16) _Float16 Bh[2][BN * BSTR];
  const int bh = blockIdx.z, b = bh >> 4, h = bh & 15;
  const float*    A  = attn + (long)bh * SEQ * SEQ;        // [SEQ, SEQ] f32
  const _Float16* Bt = vT   + (long)bh * HDIM * SEQ;       // [64, SEQ]  f16
  _Float16*       C  = ctxh + (long)b * SEQ * HIDDEN + h * HDIM;

  const int tid = threadIdx.x, lane = tid & 31, wid = tid >> 5;
  const int wm = wid * 32;
  const int m0 = blockIdx.y * BM;                           // n0 = 0 (N == 64)

  v8f acc[2][4] = {};
  stage_async_h<BN>(Bt, SEQ, 0, 0, Bh[0], tid);             // async B prefetch
  stage_cvt_a128(A, SEQ, m0, 0, Ah[0], tid);                // f32->f16 A tile
  int cur = 0;
  for (int k0 = 0; k0 < SEQ; k0 += BK) {
    __syncthreads();
    if (k0 + BK < SEQ) {
      stage_async_h<BN>(Bt, SEQ, 0, k0 + BK, Bh[cur ^ 1], tid);
      stage_cvt_a128(A, SEQ, m0, k0 + BK, Ah[cur ^ 1], tid);
      wait_async_le<2>();
    } else {
      wait_async_le<0>();
    }
    __syncthreads();                                        // also waits DScnt stores
    mma_2x4(Ah[cur], Bh[cur], wm, lane, acc);
    cur ^= 1;
  }

  const int lane15 = lane & 15;
  const int rbase  = m0 + wm + ((lane & 16) ? 8 : 0);
#pragma unroll
  for (int i = 0; i < 2; ++i)
#pragma unroll
    for (int r = 0; r < 8; ++r) {
      long row = rbase + i * 16 + r;
#pragma unroll
      for (int j = 0; j < 4; ++j)
        C[row * HIDDEN + j * 16 + lane15] = (_Float16)acc[i][j][r];
    }
}

// ===== elementwise f32 -> f16 =====
__global__ __launch_bounds__(256) void cvt_kernel(const float* __restrict__ s,
                                                  _Float16* __restrict__ d) {
  long i = ((long)blockIdx.x * 256 + threadIdx.x) * 4;
  const float4 v = *reinterpret_cast<const float4*>(s + i);
  d[i + 0] = (_Float16)v.x; d[i + 1] = (_Float16)v.y;
  d[i + 2] = (_Float16)v.z; d[i + 3] = (_Float16)v.w;
}

// ===== transpose + convert: W[R,C] f32 -> Wt[C,R] f16 (R,C multiples of 32) =====
__global__ __launch_bounds__(256) void transpose_cvt_kernel(
    const float* __restrict__ src, _Float16* __restrict__ dst, int R, int C) {
  __shared__ float t[32][33];
  const int bx = blockIdx.x * 32, by = blockIdx.y * 32;     // bx: col, by: row
  const int tx = threadIdx.x & 31, ty = threadIdx.x >> 5;   // 32 x 8
#pragma unroll
  for (int j = 0; j < 32; j += 8)
    t[ty + j][tx] = src[(long)(by + ty + j) * C + bx + tx];
  __syncthreads();
#pragma unroll
  for (int j = 0; j < 32; j += 8)
    dst[(long)(bx + ty + j) * R + by + tx] = (_Float16)t[tx][ty + j];
}

// ===== V [B,L,H,D] f32 -> V^T [B,H,D,L] f16 =====
__global__ __launch_bounds__(256) void vtrans_kernel(const float* __restrict__ vb,
                                                     _Float16* __restrict__ vT) {
  __shared__ float t[32][33];
  const int bh = blockIdx.z, b = bh >> 4, h = bh & 15;
  const float* src = vb + (long)b * SEQ * HIDDEN + h * HDIM;  // [SEQ,64] stride HIDDEN
  _Float16*    dst = vT + (long)bh * HDIM * SEQ;              // [64, SEQ]
  const int d0 = blockIdx.x * 32, l0 = blockIdx.y * 32;
  const int tx = threadIdx.x & 31, ty = threadIdx.x >> 5;
#pragma unroll
  for (int j = 0; j < 32; j += 8)
    t[ty + j][tx] = src[(long)(l0 + ty + j) * HIDDEN + d0 + tx];
  __syncthreads();
#pragma unroll
  for (int j = 0; j < 32; j += 8)
    dst[(long)(d0 + ty + j) * SEQ + l0 + tx] = (_Float16)t[tx][ty + j];
}

// ===== wesr tail + tanh: h1 = tanh(h1 + wesr @ Wm1[1024:1031]) =====
__global__ __launch_bounds__(256) void wesr_tanh_kernel(
    float* __restrict__ h1, const float* __restrict__ wesr,
    const float* __restrict__ Wm1tail) {
  long idx = (long)blockIdx.x * 256 + threadIdx.x;
  long row = idx >> 10;
  int  col = (int)(idx & 1023);
  float v = h1[idx];
#pragma unroll
  for (int j = 0; j < WESR; ++j)
    v += wesr[row * WESR + j] * Wm1tail[j * HIDDEN + col];
  h1[idx] = tanhf(v);
}

// ===== gate = sigmoid(h1 @ Wm2 + bm2) ; Wm2 [1024,16] =====
__global__ __launch_bounds__(256) void gate_kernel(
    const float* __restrict__ h1, const float* __restrict__ Wm2,
    const float* __restrict__ bm2, float* __restrict__ gate) {
  long row = blockIdx.x;
  __shared__ float hrow[HIDDEN];
  __shared__ float red[HEADS][17];
  const int t = threadIdx.x;
  for (int i = t; i < HIDDEN; i += 256) hrow[i] = h1[row * HIDDEN + i];
  __syncthreads();
  const int head = t & 15, chunk = t >> 4;
  float p = 0.0f;
#pragma unroll 8
  for (int i = chunk * 64; i < chunk * 64 + 64; ++i) p += hrow[i] * Wm2[i * HEADS + head];
  red[head][chunk] = p;
  __syncthreads();
  if (t < HEADS) {
    float s = bm2[t];
#pragma unroll
    for (int c = 0; c < 16; ++c) s += red[t][c];
    gate[row * HEADS + t] = 1.0f / (1.0f + __expf(-s));
  }
}

// ===== row softmax in place over last dim (SEQ) =====
__global__ __launch_bounds__(256) void softmax_kernel(float* __restrict__ attn) {
  long row = blockIdx.x;
  float* p = attn + row * (long)SEQ;
  const int t = threadIdx.x;
  float vals[8];
  float mx = -INFINITY;
#pragma unroll
  for (int i = 0; i < 8; ++i) { vals[i] = p[t + i * 256]; mx = fmaxf(mx, vals[i]); }
  __shared__ float red[256];
  red[t] = mx; __syncthreads();
  for (int s = 128; s > 0; s >>= 1) { if (t < s) red[t] = fmaxf(red[t], red[t + s]); __syncthreads(); }
  mx = red[0]; __syncthreads();
  float sum = 0.0f;
#pragma unroll
  for (int i = 0; i < 8; ++i) { vals[i] = __expf(vals[i] - mx); sum += vals[i]; }
  red[t] = sum; __syncthreads();
  for (int s = 128; s > 0; s >>= 1) { if (t < s) red[t] += red[t + s]; __syncthreads(); }
  float inv = 1.0f / red[0];
#pragma unroll
  for (int i = 0; i < 8; ++i) p[t + i * 256] = vals[i] * inv;
}

// ================= host-side launch =================
extern "C" void kernel_launch(void* const* d_in, const int* in_sizes, int n_in,
                              void* d_out, int out_size, void* d_ws, size_t ws_size,
                              hipStream_t stream) {
  (void)in_sizes; (void)n_in; (void)out_size; (void)ws_size;
  const float* seq   = (const float*)d_in[0];
  const int*   amask = (const int*)  d_in[1];
  const float* wesr  = (const float*)d_in[2];
  const float* Wq = (const float*)d_in[3];  const float* bq = (const float*)d_in[4];
  const float* Wk = (const float*)d_in[5];  const float* bk = (const float*)d_in[6];
  const float* Wv = (const float*)d_in[7];  const float* bv = (const float*)d_in[8];
  const float* Wo = (const float*)d_in[9];  const float* bo = (const float*)d_in[10];
  const float* Wm1 = (const float*)d_in[11]; const float* bm1 = (const float*)d_in[12];
  const float* Wm2 = (const float*)d_in[13]; const float* bm2 = (const float*)d_in[14];

  float* out  = (float*)d_out;                         // [B, L, HIDDEN]
  float* attn = out + (size_t)ROWSZ * HIDDEN;          // [B, H, L, L]

  // ---- workspace carve-up ----
  char* w = (char*)d_ws;
  const size_t NH = (size_t)ROWSZ * HIDDEN;            // 4M elements
  float*    vb    = (float*)w;               w += NH * 4;                 // f32 V
  float*    h1    = (float*)w;               w += NH * 4;                 // f32 MLP hidden
  float*    gate  = (float*)w;               w += (size_t)ROWSZ * HEADS * 4;
  _Float16* seqh  = (_Float16*)w;            w += NH * 2;
  _Float16* qh    = (_Float16*)w;            w += NH * 2;
  _Float16* kh    = (_Float16*)w;            w += NH * 2;
  _Float16* vT    = (_Float16*)w;            w += NH * 2;                 // [B,H,D,L]
  _Float16* ctxh  = (_Float16*)w;            w += NH * 2;
  const size_t WSZ = (size_t)HIDDEN * HIDDEN;
  _Float16* Wqt  = (_Float16*)w;             w += WSZ * 2;
  _Float16* Wkt  = (_Float16*)w;             w += WSZ * 2;
  _Float16* Wvt  = (_Float16*)w;             w += WSZ * 2;
  _Float16* Wot  = (_Float16*)w;             w += WSZ * 2;
  _Float16* Wm1t = (_Float16*)w;             w += WSZ * 2;

  dim3 blk(TPB);
  dim3 gProj(HIDDEN / BN, ROWSZ / BM, 1);              // 16 x 32
  dim3 gT(HIDDEN / 32, HIDDEN / 32, 1);

  // f16 conversions (once per launch)
  cvt_kernel<<<NH / (256 * 4), 256, 0, stream>>>(seq, seqh);
  transpose_cvt_kernel<<<gT, 256, 0, stream>>>(Wq,  Wqt,  HIDDEN, HIDDEN);
  transpose_cvt_kernel<<<gT, 256, 0, stream>>>(Wk,  Wkt,  HIDDEN, HIDDEN);
  transpose_cvt_kernel<<<gT, 256, 0, stream>>>(Wv,  Wvt,  HIDDEN, HIDDEN);
  transpose_cvt_kernel<<<gT, 256, 0, stream>>>(Wo,  Wot,  HIDDEN, HIDDEN);
  transpose_cvt_kernel<<<gT, 256, 0, stream>>>(Wm1, Wm1t, HIDDEN, HIDDEN); // first 1024 rows

  // projections (Q/K in f16 directly; V and MLP hidden in f32)
  gemm_nt_kernel<true ><<<gProj, blk, 0, stream>>>(seqh, HIDDEN, Wqt,  HIDDEN, bq,  qh, HIDDEN, HIDDEN);
  gemm_nt_kernel<true ><<<gProj, blk, 0, stream>>>(seqh, HIDDEN, Wkt,  HIDDEN, bk,  kh, HIDDEN, HIDDEN);
  gemm_nt_kernel<false><<<gProj, blk, 0, stream>>>(seqh, HIDDEN, Wvt,  HIDDEN, bv,  vb, HIDDEN, HIDDEN);
  gemm_nt_kernel<false><<<gProj, blk, 0, stream>>>(seqh, HIDDEN, Wm1t, HIDDEN, bm1, h1, HIDDEN, HIDDEN);

  // gate path
  wesr_tanh_kernel<<<(ROWSZ * HIDDEN) / 256, 256, 0, stream>>>(h1, wesr, Wm1 + (size_t)HIDDEN * HIDDEN);
  gate_kernel<<<ROWSZ, 256, 0, stream>>>(h1, Wm2, bm2, gate);

  // V^T for attn@V
  vtrans_kernel<<<dim3(HDIM / 32, SEQ / 32, BATCH * HEADS), 256, 0, stream>>>(vb, vT);

  // attention scores -> attn output region (pre-softmax), then softmax in place
  dim3 gS(SEQ / BN, SEQ / BM, BATCH * HEADS);          // 32 x 16 x 32
  scores_kernel<<<gS, blk, 0, stream>>>(qh, kh, gate, amask, attn);
  softmax_kernel<<<BATCH * HEADS * SEQ, 256, 0, stream>>>(attn);

  // attn @ V -> ctx (f16), then output projection
  dim3 gAV(1, SEQ / BM, BATCH * HEADS);
  av_kernel<<<gAV, blk, 0, stream>>>(attn, vT, ctxh);
  gemm_nt_kernel<false><<<gProj, blk, 0, stream>>>(ctxh, HIDDEN, Wot, HIDDEN, bo, out, HIDDEN, HIDDEN);
}
